// GNN_58858231824789
// MI455X (gfx1250) — matrix operand order repaired and checked by reading, not compiled, not run on hardware
//
#include <hip/hip_runtime.h>
#include <hip/hip_bf16.h>

typedef __attribute__((ext_vector_type(16))) __bf16 v16bf;
typedef __attribute__((ext_vector_type(8)))  __bf16 v8bf;
typedef __attribute__((ext_vector_type(8)))  float  v8f;

#define H 128
#define LAYERS 4

// ---------------------------------------------------------------------------
// CDNA5 async global->LDS copy (ASYNCcnt-tracked, no VGPR data round-trip)
// ---------------------------------------------------------------------------
__device__ __forceinline__ void async_copy16(unsigned lds_off, const void* gptr) {
    asm volatile("global_load_async_to_lds_b128 %0, %1, off"
                 :: "v"(lds_off), "v"((unsigned long long)gptr)
                 : "memory");
}
__device__ __forceinline__ void wait_async() {
    asm volatile("s_wait_asynccnt 0x0" ::: "memory");
}

// ---------------------------------------------------------------------------
// WMMA helpers (wave32, 16x16x32 bf16, fp32 accumulate, K=128 in 4 steps)
// ---------------------------------------------------------------------------

// A-matrix 16x32 bf16 fragment per ISA layout:
//   lanes 0-15 : row M=lane,    elems 0..7 = K c0..c0+7, 8..15 = K c0+16..+23 (c0=0)
//   lanes 16-31: row M=lane-16, same with c0=8
__device__ __forceinline__ void load_afrag_global(v16bf A[4], const float* __restrict__ rowptr, int lane) {
    const int c0 = (lane < 16) ? 0 : 8;
#pragma unroll
    for (int s = 0; s < 4; ++s) {
        const float* p0 = rowptr + 32 * s + c0;
        const float* p1 = p0 + 16;
        float4 a0 = *(const float4*)(p0);
        float4 a1 = *(const float4*)(p0 + 4);
        float4 b0 = *(const float4*)(p1);
        float4 b1 = *(const float4*)(p1 + 4);
        v16bf a;
        a[0] = (__bf16)a0.x;  a[1] = (__bf16)a0.y;  a[2]  = (__bf16)a0.z;  a[3]  = (__bf16)a0.w;
        a[4] = (__bf16)a1.x;  a[5] = (__bf16)a1.y;  a[6]  = (__bf16)a1.z;  a[7]  = (__bf16)a1.w;
        a[8] = (__bf16)b0.x;  a[9] = (__bf16)b0.y;  a[10] = (__bf16)b0.z;  a[11] = (__bf16)b0.w;
        a[12] = (__bf16)b1.x; a[13] = (__bf16)b1.y; a[14] = (__bf16)b1.z;  a[15] = (__bf16)b1.w;
        A[s] = a;
    }
}

__device__ __forceinline__ void load_afrag_lds(v16bf A[4], const __bf16* buf, int lane) {
    const int row = lane & 15;
    const int c0  = (lane < 16) ? 0 : 8;
    const __bf16* rp = buf + row * H;
#pragma unroll
    for (int s = 0; s < 4; ++s) {
        const __bf16* p0 = rp + 32 * s + c0;
        v8bf q0 = *(const v8bf*)(p0);
        v8bf q1 = *(const v8bf*)(p0 + 16);
        v16bf a;
#pragma unroll
        for (int e = 0; e < 8; ++e) { a[e] = q0[e]; a[8 + e] = q1[e]; }
        A[s] = a;
    }
}

// B-matrix 32x16 bf16: lane n (0-15) holds column n, elems 0..7 = K 0..7,
// 8..15 = K 16..23; lanes 16-31 same with K+8.  W row-major [O,128]: each lane
// reads two 8-bf16 chunks from W[row = o_tile*16 + (lane&15)].  Works for global or LDS W.
__device__ __forceinline__ v8f gemm_tile(const v16bf A[4], const __bf16* W,
                                         int o_tile, int lane) {
    const int wrow  = o_tile * 16 + (lane & 15);
    const int kbase = (lane < 16) ? 0 : 8;
    const __bf16* wp = W + wrow * H + kbase;
    v8f c = {};
#pragma unroll
    for (int s = 0; s < 4; ++s) {
        v8bf q0 = *(const v8bf*)(wp + 32 * s);
        v8bf q1 = *(const v8bf*)(wp + 32 * s + 16);
        v16bf b;
#pragma unroll
        for (int e = 0; e < 8; ++e) { b[e] = q0[e]; b[8 + e] = q1[e]; }
        c = __builtin_amdgcn_wmma_f32_16x16x32_bf16(false, A[s], false, b,
                                                    (short)0, c, false, false);
    }
    return c;
}

__device__ __forceinline__ float sigmoidf_fast(float x) {
    return 1.0f / (1.0f + __expf(-x));
}

// ---------------------------------------------------------------------------
// Small utility kernels
// ---------------------------------------------------------------------------
__global__ void cvt_bf16_kernel(const float* __restrict__ src, __bf16* __restrict__ dst, int n) {
    int i = blockIdx.x * blockDim.x + threadIdx.x;
    if (i < n) dst[i] = (__bf16)src[i];
}

__global__ void zero_f32_kernel(float* __restrict__ p, int n) {
    int i = blockIdx.x * blockDim.x + threadIdx.x;
    int stride = gridDim.x * blockDim.x;
    for (; i < n; i += stride) p[i] = 0.0f;
}

// ---------------------------------------------------------------------------
// Edge scatter: h[dst] += x[src].  One wave per edge, float4 per lane.
// x and h (51 MB each) are L2-resident on the 192MB L2; atomics resolve there.
// ---------------------------------------------------------------------------
__global__ void __launch_bounds__(256)
scatter_kernel(const int* __restrict__ ei, const float* __restrict__ x,
               float* __restrict__ h, int E) {
    const int lane = threadIdx.x & 31;
    const int gw   = (blockIdx.x * blockDim.x + threadIdx.x) >> 5;
    const int nw   = (gridDim.x * blockDim.x) >> 5;
    for (int e = gw; e < E; e += nw) {
        const int src = ei[e];
        const int dst = ei[E + e];
        float4 v = ((const float4*)(x + (size_t)src * H))[lane];
        float* hd = h + (size_t)dst * H + lane * 4;
        atomicAdd(hd + 0, v.x);
        atomicAdd(hd + 1, v.y);
        atomicAdd(hd + 2, v.z);
        atomicAdd(hd + 3, v.w);
    }
}

// ---------------------------------------------------------------------------
// Preprocessor linear: Y = X @ W^T + b  (128x128, bf16 WMMA, fp32 out).
// One small GEMM; weights read straight from L2.
// ---------------------------------------------------------------------------
__global__ void __launch_bounds__(256)
pre_gemm_kernel(const float* __restrict__ X, const __bf16* __restrict__ W,
                const float* __restrict__ bias, float* __restrict__ Y, int N) {
    const int lane = threadIdx.x & 31;
    const int wave = threadIdx.x >> 5;
    const long base = ((long)blockIdx.x * 8 + wave) * 16;
    if (base >= N) return;
    const int mrow  = lane & 15;
    const int mbase = (lane >> 4) << 3;

    v16bf Ax[4];
    load_afrag_global(Ax, X + (base + mrow) * H, lane);

#pragma unroll
    for (int j = 0; j < 8; ++j) {
        v8f c = gemm_tile(Ax, W, j, lane);
        const int col = j * 16 + mrow;
        const float bv = bias[col];
#pragma unroll
        for (int v = 0; v < 8; ++v)
            Y[(base + mbase + v) * H + col] = c[v] + bv;
    }
}

// ---------------------------------------------------------------------------
// Fused layer: GRU cell + 2-layer MLP + skip + LayerNorm + ReLU.
// 256 threads = 8 waves, each wave owns a 16-row tile (128 rows/block).
// Weights are staged into LDS ONCE PER BLOCK with async global->LDS copies
// (8x less L2 traffic than per-wave streaming), WMMA B-fragments read from
// LDS.  Intermediate 16x128 activations round-trip a private 4KB/wave LDS
// buffer as bf16 (same-wave LDS ops are in-order; no barrier needed there).
// ---------------------------------------------------------------------------
__global__ void __launch_bounds__(256)
fused_layer_kernel(const float* __restrict__ xin, const float* __restrict__ h,
                   const float* __restrict__ x0in,
                   const __bf16* __restrict__ Wih, const __bf16* __restrict__ Whh,
                   const __bf16* __restrict__ W1,  const __bf16* __restrict__ W2,
                   const float* __restrict__ bih,  const float* __restrict__ bhh,
                   const float* __restrict__ b1,   const float* __restrict__ b2,
                   const float* __restrict__ ln_g, const float* __restrict__ ln_b,
                   float* __restrict__ x0out, float* __restrict__ post, int N) {
    __shared__ __bf16 sW[16384];        // 32 KB weight staging (up to 8 16x128 tiles)
    __shared__ __bf16 sbuf[8][16 * H];  // 32 KB per-wave activation buffers

    const int tid  = threadIdx.x;
    const int lane = tid & 31;
    const int wave = tid >> 5;
    const long base = ((long)blockIdx.x * 8 + wave) * 16;
    const bool active = (base < (long)N);
    const int mrow  = lane & 15;
    const int mbase = (lane >> 4) << 3;
    __bf16* sb = &sbuf[wave][0];
    const unsigned sW_off = (unsigned)(unsigned long long)(const void*)&sW[0];

    // ---- load per-wave A fragments ------------------------------------
    v16bf Ax[4], Ah[4];
    if (active) {
        load_afrag_global(Ax, xin + (base + mrow) * H, lane);
        load_afrag_global(Ah, h   + (base + mrow) * H, lane);
    }

    // ---- GRU gates [r,z,n]; out = (1-z)*n + z*h -----------------------
    // Per output-column tile j, stage the six 16x128 gate weight tiles
    // (Wih r/z/n, Whh r/z/n rows j*16..j*16+15) = 24 KB, then all waves
    // consume them from LDS.
    for (int j = 0; j < 8; ++j) {
        __syncthreads();                       // previous j's tiles consumed
        {
            const __bf16* gsrc[6] = {
                Wih + (size_t)(0 * H + j * 16) * H,
                Wih + (size_t)(1 * H + j * 16) * H,
                Wih + (size_t)(2 * H + j * 16) * H,
                Whh + (size_t)(0 * H + j * 16) * H,
                Whh + (size_t)(1 * H + j * 16) * H,
                Whh + (size_t)(2 * H + j * 16) * H };
#pragma unroll
            for (int t = 0; t < 6; ++t)        // 6 tiles x 4KB, 16B per thread each
                async_copy16(sW_off + t * 4096 + tid * 16,
                             (const char*)gsrc[t] + tid * 16);
        }
        wait_async();
        __syncthreads();                       // staged tiles visible to all waves

        if (active) {
            v8f ir  = gemm_tile(Ax, sW + 0 * 2048, 0, lane);
            v8f iz  = gemm_tile(Ax, sW + 1 * 2048, 0, lane);
            v8f in_ = gemm_tile(Ax, sW + 2 * 2048, 0, lane);
            v8f hr  = gemm_tile(Ah, sW + 3 * 2048, 0, lane);
            v8f hz  = gemm_tile(Ah, sW + 4 * 2048, 0, lane);
            v8f hn  = gemm_tile(Ah, sW + 5 * 2048, 0, lane);
            const int col = j * 16 + mrow;
            const float bir = bih[col],         bhr = bhh[col];
            const float biz = bih[H + col],     bhz = bhh[H + col];
            const float bin = bih[2 * H + col], bhn = bhh[2 * H + col];
#pragma unroll
            for (int v = 0; v < 8; ++v) {
                const long row = base + mbase + v;
                float r = sigmoidf_fast(ir[v] + bir + hr[v] + bhr);
                float z = sigmoidf_fast(iz[v] + biz + hz[v] + bhz);
                float n = tanhf(in_[v] + bin + r * (hn[v] + bhn));
                float hv = h[row * H + col];
                float o = (1.0f - z) * n + z * hv;
                sb[(mbase + v) * H + col] = (__bf16)o;   // C-layout -> row-major LDS
            }
        }
    }

    // ---- t = relu(out @ W1^T + b1): stage full W1 (32 KB) once --------
    __syncthreads();
    for (int c = tid; c < 2048; c += 256)
        async_copy16(sW_off + c * 16, (const char*)W1 + c * 16);
    wait_async();
    __syncthreads();

    v16bf Ao[4];
    if (active) {
        load_afrag_lds(Ao, sb, lane);          // whole out tile now in regs
        for (int j = 0; j < 8; ++j) {
            v8f acc = gemm_tile(Ao, sW, j, lane);
            const int col = j * 16 + mrow;
            const float bv = b1[col];
#pragma unroll
            for (int v = 0; v < 8; ++v) {
                float t = fmaxf(acc[v] + bv, 0.0f);
                sb[(mbase + v) * H + col] = (__bf16)t;   // reuse buffer (Ao in regs)
            }
        }
    }

    // ---- z2 = t @ W2^T + b2: stage full W2 (32 KB) once ---------------
    __syncthreads();
    for (int c = tid; c < 2048; c += 256)
        async_copy16(sW_off + c * 16, (const char*)W2 + c * 16);
    wait_async();
    __syncthreads();

    if (!active) return;

    v16bf At[4];
    load_afrag_lds(At, sb, lane);
    float xn[8][8];
    float s1[8], s2[8];
#pragma unroll
    for (int v = 0; v < 8; ++v) { s1[v] = 0.0f; s2[v] = 0.0f; }

#pragma unroll
    for (int j = 0; j < 8; ++j) {
        v8f acc = gemm_tile(At, sW, j, lane);
        const int col = j * 16 + mrow;
        const float bv = b2[col];
#pragma unroll
        for (int v = 0; v < 8; ++v) {
            const long row = base + mbase + v;
            float val = x0in[row * H + col] + acc[v] + bv;
            x0out[row * H + col] = val;              // new skip state
            xn[j][v] = val;
            s1[v] += val;
            s2[v] += val * val;
        }
    }

    // LayerNorm stats: row sums live across the 16 lanes of each C half.
    float mu[8], rs[8];
#pragma unroll
    for (int v = 0; v < 8; ++v) {
        float a = s1[v], b = s2[v];
        for (int m = 1; m < 16; m <<= 1) {
            a += __shfl_xor(a, m);
            b += __shfl_xor(b, m);
        }
        const float mean = a * (1.0f / (float)H);
        const float var  = b * (1.0f / (float)H) - mean * mean;
        mu[v] = mean;
        rs[v] = rsqrtf(var + 1e-5f);
    }

    // ---- post = relu(layernorm(xn)) -----------------------------------
#pragma unroll
    for (int j = 0; j < 8; ++j) {
        const int col = j * 16 + mrow;
        const float g = ln_g[col], bb = ln_b[col];
#pragma unroll
        for (int v = 0; v < 8; ++v) {
            const long row = base + mbase + v;
            float o = g * (xn[j][v] - mu[v]) * rs[v] + bb;
            post[row * H + col] = fmaxf(o, 0.0f);
        }
    }
}

// ---------------------------------------------------------------------------
// Host-side launch
// ---------------------------------------------------------------------------
extern "C" void kernel_launch(void* const* d_in, const int* in_sizes, int n_in,
                              void* d_out, int out_size, void* d_ws, size_t ws_size,
                              hipStream_t stream) {
    const float* x     = (const float*)d_in[0];
    const int*   ei    = (const int*)d_in[1];
    const float* pre_W = (const float*)d_in[2];
    const float* pre_b = (const float*)d_in[3];
    const float* Wih   = (const float*)d_in[4];
    const float* Whh   = (const float*)d_in[5];
    const float* bih   = (const float*)d_in[6];
    const float* bhh   = (const float*)d_in[7];
    const float* W1    = (const float*)d_in[8];
    const float* b1    = (const float*)d_in[9];
    const float* W2    = (const float*)d_in[10];
    const float* b2    = (const float*)d_in[11];
    const float* ln_g  = (const float*)d_in[12];
    const float* ln_b  = (const float*)d_in[13];
    float* out = (float*)d_out;

    const int N = in_sizes[0] / H;       // 100000
    const int E = in_sizes[1] / 2;       // 1600000
    const size_t NH = (size_t)N * H;

    // workspace: [h | x0 | xpre | bf16 weights]
    float* hbuf  = (float*)d_ws;
    float* x0buf = hbuf + NH;
    float* xpre  = x0buf + NH;
    __bf16* bfb  = (__bf16*)(xpre + NH);
    __bf16* preWb = bfb;
    __bf16* Wihb  = preWb + 128 * 128;
    __bf16* Whhb  = Wihb + (size_t)LAYERS * 384 * 128;
    __bf16* W1b   = Whhb + (size_t)LAYERS * 384 * 128;
    __bf16* W2b   = W1b  + (size_t)LAYERS * 128 * 128;

    // weights -> bf16 (recomputed every call; deterministic)
    {
        int n;
        n = 128 * 128;          cvt_bf16_kernel<<<(n + 255) / 256, 256, 0, stream>>>(pre_W, preWb, n);
        n = LAYERS * 384 * 128; cvt_bf16_kernel<<<(n + 255) / 256, 256, 0, stream>>>(Wih, Wihb, n);
        n = LAYERS * 384 * 128; cvt_bf16_kernel<<<(n + 255) / 256, 256, 0, stream>>>(Whh, Whhb, n);
        n = LAYERS * 128 * 128; cvt_bf16_kernel<<<(n + 255) / 256, 256, 0, stream>>>(W1, W1b, n);
        n = LAYERS * 128 * 128; cvt_bf16_kernel<<<(n + 255) / 256, 256, 0, stream>>>(W2, W2b, n);
    }

    const int tiles  = (N + 15) / 16;
    const int blocks = (tiles + 7) / 8;

    pre_gemm_kernel<<<blocks, 256, 0, stream>>>(x, preWb, pre_b, xpre, N);

    for (int i = 0; i < LAYERS; ++i) {
        zero_f32_kernel<<<1024, 256, 0, stream>>>(hbuf, (int)NH);

        const float* xin  = (i == 0) ? xpre : out + (size_t)(i - 1) * NH;
        scatter_kernel<<<2048, 256, 0, stream>>>(ei, xin, hbuf, E);

        const float* x0in = (i == 0) ? xpre : x0buf;
        fused_layer_kernel<<<blocks, 256, 0, stream>>>(
            xin, hbuf, x0in,
            Wihb + (size_t)i * 384 * 128, Whhb + (size_t)i * 384 * 128,
            W1b + (size_t)i * 128 * 128,  W2b + (size_t)i * 128 * 128,
            bih + (size_t)i * 384, bhh + (size_t)i * 384,
            b1 + (size_t)i * 128,  b2 + (size_t)i * 128,
            ln_g, ln_b, x0buf, out + (size_t)i * NH, N);
    }
}
